// MECGNN_86912958202308
// MI455X (gfx1250) — matrix operand order repaired and checked by reading, not compiled, not run on hardware
//
#include <hip/hip_runtime.h>
#include <hip/hip_bf16.h>

#define N_NODES 25000
#define N_EDGES 400000
#define HIDDIM  128
#define LAYERS  3
#define OUT_DIM 10
#define LN_EPS  1e-5f
#define EPB     64              // edges per edgeconv block (4 M-tiles of 16)

typedef _Float16 half16 __attribute__((ext_vector_type(16)));
typedef float    float8 __attribute__((ext_vector_type(8)));
typedef int      v4i    __attribute__((ext_vector_type(4)));
typedef v4i __attribute__((address_space(1))) gv4i;   // global
typedef v4i __attribute__((address_space(3))) lv4i;   // LDS

__device__ __forceinline__ float lrelu(float x, float s) { return x >= 0.f ? x : s * x; }
__device__ __forceinline__ unsigned fenc(float f) {
    unsigned u = __float_as_uint(f);
    return (u & 0x80000000u) ? ~u : (u | 0x80000000u);
}
__device__ __forceinline__ float fdec(unsigned e) {
    unsigned u = (e & 0x80000000u) ? (e & 0x7FFFFFFFu) : ~e;
    return __uint_as_float(u);
}

// --- CDNA5 async global->LDS copy (16B per call), with portable fallback ----
__device__ __forceinline__ void async_copy16(void* lds_dst, const void* gsrc)
{
#if __has_builtin(__builtin_amdgcn_global_load_async_to_lds_b128)
    __builtin_amdgcn_global_load_async_to_lds_b128(
        (gv4i*)(unsigned long long)gsrc,
        (lv4i*)(unsigned int)(unsigned long long)lds_dst,  // ISA 10.2: LDS = addr[31:0]
        0, 0);
#else
    *(uint4*)lds_dst = *(const uint4*)gsrc;
#endif
}
__device__ __forceinline__ void async_join()
{
#if __has_builtin(__builtin_amdgcn_s_wait_asynccnt)
    __builtin_amdgcn_s_wait_asynccnt(0);
#elif defined(__gfx1250__)
    asm volatile("s_wait_asynccnt 0x0" ::: "memory");
#endif
}
__device__ __forceinline__ void lds_bulk_copy(void* lds, const void* g, int bytes,
                                              int tid, int nthreads)
{
    for (int off = tid * 16; off < bytes; off += nthreads * 16)
        async_copy16((char*)lds + off, (const char*)g + off);
}

// ---------------------------------------------------------------------------
// Weight pre-convert + transpose: Wt[m][n*K+k] = (f16) W[m][k*N+n]
// ---------------------------------------------------------------------------
__global__ void cvt_wt_k(const float* __restrict__ W, _Float16* __restrict__ Wt,
                         int K, int N, int count)
{
    long idx = (long)blockIdx.x * 256 + threadIdx.x;
    long per = (long)K * N;
    if (idx >= per * count) return;
    int m = (int)(idx / per);
    int r = (int)(idx % per);
    int n = r / K, k = r % K;
    Wt[m * per + n * K + k] = (_Float16)W[m * per + k * N + n];
}

// ---------------------------------------------------------------------------
// Node GEMM: Y[N,128] = act(LN(X[N,128] @ W + bias)); W pre-transposed f16.
// X from f32 (Xf) or f16 shadow (Xh). Optional f16 shadow output Yh.
// ---------------------------------------------------------------------------
__global__ __launch_bounds__(256)
void node_gemm_k128(const float* __restrict__ Xf, const _Float16* __restrict__ Xh,
                    const _Float16* __restrict__ Wt_g,
                    const float* __restrict__ bias, const float* __restrict__ gamma,
                    const float* __restrict__ beta, float* __restrict__ Y,
                    _Float16* __restrict__ Yh,
                    int nrows, int do_ln, int do_act, float slope)
{
    __shared__ __align__(16) _Float16 Wt[128][128];   // Wt[n][k]
    __shared__ __align__(16) _Float16 As[16][128];
    __shared__ float stage[16][128];
    __shared__ float psum[16][8], psq[16][8];
    __shared__ float mu[16], ristd[16];

    const int tid  = threadIdx.x;
    const int lane = tid & 31;
    const int wave = tid >> 5;
    const int base = blockIdx.x * 16;
    const int lg   = lane >> 4;
    const int lm   = lane & 15;

    lds_bulk_copy(&Wt[0][0], Wt_g, 128 * 128 * 2, tid, 256);
    if (Xh) {
        int r = tid >> 4, c = tid & 15;          // 256 threads = 16 rows x 16 chunks
        int row = base + r;
        if (row < nrows)
            async_copy16((char*)&As[r][0] + c * 16,
                         (const char*)(Xh + (size_t)row * 128) + c * 16);
        else {
            uint4 z = {0, 0, 0, 0};
            *(uint4*)((char*)&As[r][0] + c * 16) = z;
        }
    } else {
        for (int idx = tid; idx < 16 * 128; idx += 256) {
            int r = idx >> 7, c = idx & 127;
            int row = base + r;
            As[r][c] = (row < nrows) ? (_Float16)Xf[(size_t)row * 128 + c] : (_Float16)0.f;
        }
    }
    async_join();
    __syncthreads();

    const int n0 = wave * 16;
    float8 acc = {0.f, 0.f, 0.f, 0.f, 0.f, 0.f, 0.f, 0.f};
    #pragma unroll
    for (int kk = 0; kk < 128; kk += 32) {
        half16 a, b;
        const _Float16* pa = &As[lm][kk + lg * 8];
        #pragma unroll
        for (int q = 0; q < 8; q++) { a[q] = pa[q]; a[q + 8] = pa[q + 16]; }
        const _Float16* pb = &Wt[n0 + lm][kk + lg * 8];
        #pragma unroll
        for (int q = 0; q < 8; q++) { b[q] = pb[q]; b[q + 8] = pb[q + 16]; }
        acc = __builtin_amdgcn_wmma_f32_16x16x32_f16(false, a, false, b,
                                                     (short)0, acc, false, false);
    }
    float bcol = bias ? bias[n0 + lm] : 0.f;

    if (do_ln) {
        #pragma unroll
        for (int j = 0; j < 8; j++) stage[j + lg * 8][n0 + lm] = acc[j] + bcol;
        __syncthreads();
        if (tid < 128) {
            int r = tid >> 3, seg = tid & 7;
            float s = 0.f, ss = 0.f;
            #pragma unroll
            for (int q = 0; q < 16; q++) { float v = stage[r][seg * 16 + q]; s += v; ss += v * v; }
            psum[r][seg] = s; psq[r][seg] = ss;
        }
        __syncthreads();
        if (tid < 16) {
            float s = 0.f, ss = 0.f;
            #pragma unroll
            for (int q = 0; q < 8; q++) { s += psum[tid][q]; ss += psq[tid][q]; }
            float m = s * (1.f / 128.f);
            float v = ss * (1.f / 128.f) - m * m;
            mu[tid] = m; ristd[tid] = rsqrtf(v + LN_EPS);
        }
        __syncthreads();
        for (int idx = tid; idx < 2048; idx += 256) {
            int r = idx >> 7, c = idx & 127;
            int row = base + r;
            if (row < nrows) {
                float v = (stage[r][c] - mu[r]) * ristd[r] * gamma[c] + beta[c];
                if (do_act) v = lrelu(v, slope);
                Y[(size_t)row * 128 + c] = v;
                if (Yh) Yh[(size_t)row * 128 + c] = (_Float16)v;
            }
        }
    } else {
        #pragma unroll
        for (int j = 0; j < 8; j++) {
            int row = base + j + lg * 8;
            if (row < nrows) {
                float v = acc[j] + bcol;
                if (do_act) v = lrelu(v, slope);
                Y[(size_t)row * 128 + n0 + lm] = v;
            }
        }
    }
}

// ---------------------------------------------------------------------------
// Fused EdgeConv, 64 edges/block, weights resident in LDS across 4 M-tiles.
// ---------------------------------------------------------------------------
__global__ __launch_bounds__(256)
void edgeconv_k(const _Float16* __restrict__ hh, const int* __restrict__ src,
                const int* __restrict__ dst, const float* __restrict__ eattr,
                const _Float16* __restrict__ W1t_g, const float* __restrict__ b1,
                const float* __restrict__ g1, const float* __restrict__ be1,
                const _Float16* __restrict__ W2t_g, const float* __restrict__ b2,
                const float* __restrict__ g2, const float* __restrict__ be2,
                float* __restrict__ x2)
{
    __shared__ __align__(16) _Float16 W1t[128][256];   // W1t[n][k]
    __shared__ __align__(16) _Float16 W2t[128][128];
    __shared__ __align__(16) _Float16 A1[EPB][256];
    __shared__ __align__(16) _Float16 A2[16][128];
    __shared__ float stage[16][128];
    __shared__ float psum[16][8], psq[16][8];
    __shared__ float mu[16], ristd[16];
    __shared__ int   sidx[EPB], didx[EPB];

    const int tid  = threadIdx.x;
    const int lane = tid & 31;
    const int wave = tid >> 5;
    const int eb   = blockIdx.x * EPB;
    const int lg   = lane >> 4;
    const int lm   = lane & 15;
    const int n0   = wave * 16;

    if (tid < EPB) { sidx[tid] = src[eb + tid]; didx[tid] = dst[eb + tid]; }
    lds_bulk_copy(&W1t[0][0], W1t_g, 128 * 256 * 2, tid, 256);
    lds_bulk_copy(&W2t[0][0], W2t_g, 128 * 128 * 2, tid, 256);
    __syncthreads();                                   // sidx/didx visible

    // gather concat(h16[src], h16[dst]) rows: EPB x 32 16B-chunks
    for (int ch = tid; ch < EPB * 32; ch += 256) {
        int r = ch >> 5, c = ch & 31;
        int node = (c < 16) ? sidx[r] : didx[r];
        async_copy16((char*)&A1[r][0] + c * 16,
                     (const char*)(hh + (size_t)node * 128) + (c & 15) * 16);
    }
    async_join();
    __syncthreads();

    for (int mt = 0; mt < EPB / 16; ++mt) {
        const int rb = mt * 16;

        // ---- GEMM1: [16,256] x [256,128] ----
        float8 acc = {0.f, 0.f, 0.f, 0.f, 0.f, 0.f, 0.f, 0.f};
        #pragma unroll
        for (int kk = 0; kk < 256; kk += 32) {
            half16 a, b;
            const _Float16* pa = &A1[rb + lm][kk + lg * 8];
            #pragma unroll
            for (int q = 0; q < 8; q++) { a[q] = pa[q]; a[q + 8] = pa[q + 16]; }
            const _Float16* pb = &W1t[n0 + lm][kk + lg * 8];
            #pragma unroll
            for (int q = 0; q < 8; q++) { b[q] = pb[q]; b[q + 8] = pb[q + 16]; }
            acc = __builtin_amdgcn_wmma_f32_16x16x32_f16(false, a, false, b,
                                                         (short)0, acc, false, false);
        }
        {
            float bcol = b1[n0 + lm];
            #pragma unroll
            for (int j = 0; j < 8; j++) stage[j + lg * 8][n0 + lm] = acc[j] + bcol;
        }
        __syncthreads();
        if (tid < 128) {
            int r = tid >> 3, seg = tid & 7;
            float s = 0.f, ss = 0.f;
            #pragma unroll
            for (int q = 0; q < 16; q++) { float v = stage[r][seg * 16 + q]; s += v; ss += v * v; }
            psum[r][seg] = s; psq[r][seg] = ss;
        }
        __syncthreads();
        if (tid < 16) {
            float s = 0.f, ss = 0.f;
            #pragma unroll
            for (int q = 0; q < 8; q++) { s += psum[tid][q]; ss += psq[tid][q]; }
            float m = s * (1.f / 128.f);
            float v = ss * (1.f / 128.f) - m * m;
            mu[tid] = m; ristd[tid] = rsqrtf(v + LN_EPS);
        }
        __syncthreads();
        for (int idx = tid; idx < 2048; idx += 256) {
            int r = idx >> 7, c = idx & 127;
            float v = (stage[r][c] - mu[r]) * ristd[r] * g1[c] + be1[c];
            A2[r][c] = (_Float16)lrelu(v, 0.2f);
        }
        __syncthreads();

        // ---- GEMM2: [16,128] x [128,128] ----
        float8 acc2 = {0.f, 0.f, 0.f, 0.f, 0.f, 0.f, 0.f, 0.f};
        #pragma unroll
        for (int kk = 0; kk < 128; kk += 32) {
            half16 a, b;
            const _Float16* pa = &A2[lm][kk + lg * 8];
            #pragma unroll
            for (int q = 0; q < 8; q++) { a[q] = pa[q]; a[q + 8] = pa[q + 16]; }
            const _Float16* pb = &W2t[n0 + lm][kk + lg * 8];
            #pragma unroll
            for (int q = 0; q < 8; q++) { b[q] = pb[q]; b[q + 8] = pb[q + 16]; }
            acc2 = __builtin_amdgcn_wmma_f32_16x16x32_f16(false, a, false, b,
                                                          (short)0, acc2, false, false);
        }
        __syncthreads();                 // A2 fully consumed before restaging
        {
            float bcol = b2[n0 + lm];
            #pragma unroll
            for (int j = 0; j < 8; j++) stage[j + lg * 8][n0 + lm] = acc2[j] + bcol;
        }
        __syncthreads();
        if (tid < 128) {
            int r = tid >> 3, seg = tid & 7;
            float s = 0.f, ss = 0.f;
            #pragma unroll
            for (int q = 0; q < 16; q++) { float v = stage[r][seg * 16 + q]; s += v; ss += v * v; }
            psum[r][seg] = s; psq[r][seg] = ss;
        }
        __syncthreads();
        if (tid < 16) {
            float s = 0.f, ss = 0.f;
            #pragma unroll
            for (int q = 0; q < 8; q++) { s += psum[tid][q]; ss += psq[tid][q]; }
            float m = s * (1.f / 128.f);
            float v = ss * (1.f / 128.f) - m * m;
            mu[tid] = m; ristd[tid] = rsqrtf(v + LN_EPS);
        }
        __syncthreads();
        for (int idx = tid; idx < 2048; idx += 256) {
            int r = idx >> 7, c = idx & 127;
            float v = (stage[r][c] - mu[r]) * ristd[r] * g2[c] + be2[c];
            v = lrelu(v, 0.2f) * eattr[eb + rb + r];
            atomicAdd(&x2[(size_t)didx[rb + r] * 128 + c], v);
        }
        __syncthreads();                 // stage/psum reused next M-tile
    }
}

// ---------------------------------------------------------------------------
// GAT attention helpers
// ---------------------------------------------------------------------------
__global__ __launch_bounds__(128)
void gat_prep_k(const float* __restrict__ ht, const float* __restrict__ as_,
                const float* __restrict__ ad_, const float* __restrict__ bias,
                float* a_s, float* a_d, float* e_self, unsigned* emax,
                float* denom, float* x1, float* x2)
{
    __shared__ float rs[128], rd[128];
    int n = blockIdx.x, c = threadIdx.x;
    float v = ht[(size_t)n * 128 + c];
    int head = c >> 6, cc = c & 63;
    rs[c] = v * as_[head * 64 + cc];
    rd[c] = v * ad_[head * 64 + cc];
    __syncthreads();
    for (int off = 32; off > 0; off >>= 1) {
        if ((c & 63) < off) { rs[c] += rs[c + off]; rd[c] += rd[c + off]; }
        __syncthreads();
    }
    if ((c & 63) == 0) {
        int hd = c >> 6;
        float s = rs[c], d = rd[c];
        a_s[n * 2 + hd] = s; a_d[n * 2 + hd] = d;
        float e = lrelu(s + d, 0.2f);
        e_self[n * 2 + hd] = e;
        emax[n * 2 + hd] = fenc(e);
        denom[n * 2 + hd] = 0.f;
    }
    x1[(size_t)n * 128 + c] = bias[c];
    x2[(size_t)n * 128 + c] = 0.f;
}

__global__ void gat_edge_max_k(const int* __restrict__ src, const int* __restrict__ dst,
                               const float* __restrict__ a_s, const float* __restrict__ a_d,
                               float* __restrict__ e_val, unsigned* __restrict__ emax)
{
    int t = blockIdx.x * blockDim.x + threadIdx.x;
    if (t >= N_EDGES * 2) return;
    int e = t >> 1, hd = t & 1;
    float ev = lrelu(a_s[src[e] * 2 + hd] + a_d[dst[e] * 2 + hd], 0.2f);
    e_val[t] = ev;
    atomicMax(&emax[dst[e] * 2 + hd], fenc(ev));
}

__global__ void gat_edge_exp_k(const int* __restrict__ dst, const float* __restrict__ e_val,
                               const unsigned* __restrict__ emax, float* __restrict__ ee,
                               float* __restrict__ denom)
{
    int t = blockIdx.x * blockDim.x + threadIdx.x;
    if (t >= N_EDGES * 2) return;
    int e = t >> 1, hd = t & 1;
    float x = __expf(e_val[t] - fdec(emax[dst[e] * 2 + hd]));
    ee[t] = x;
    atomicAdd(&denom[dst[e] * 2 + hd], x);
}

__global__ void gat_self_k(float* __restrict__ e_self, const unsigned* __restrict__ emax,
                           float* __restrict__ denom)
{
    int t = blockIdx.x * blockDim.x + threadIdx.x;
    if (t >= N_NODES * 2) return;
    float es = __expf(e_self[t] - fdec(emax[t]));
    e_self[t] = es;
    denom[t] += es;
}

__global__ __launch_bounds__(128)
void gat_scatter_k(const int* __restrict__ src, const int* __restrict__ dst,
                   const float* __restrict__ ht, const float* __restrict__ ee,
                   const float* __restrict__ denom, float* __restrict__ x1)
{
    int e = blockIdx.x, c = threadIdx.x;
    int sN = src[e], dN = dst[e];
    int hd = c >> 6;
    float alpha = ee[e * 2 + hd] / denom[dN * 2 + hd];
    atomicAdd(&x1[(size_t)dN * 128 + c], ht[(size_t)sN * 128 + c] * alpha);
}

__global__ __launch_bounds__(128)
void combine_k(const float* __restrict__ x1, const float* __restrict__ x2,
               const float* __restrict__ ht, const float* __restrict__ es,
               const float* __restrict__ denom, const float* __restrict__ gamma,
               const float* __restrict__ beta, float* __restrict__ h,
               _Float16* __restrict__ hh, float* __restrict__ x0, int add_res)
{
    __shared__ float red[128], red2[128];
    __shared__ float mu_s, ris_s;
    int n = blockIdx.x, c = threadIdx.x;
    int hd = c >> 6;
    float self_alpha = es[n * 2 + hd] / denom[n * 2 + hd];
    float v = x1[(size_t)n * 128 + c] + x2[(size_t)n * 128 + c] +
              ht[(size_t)n * 128 + c] * self_alpha;
    red[c] = v; red2[c] = v * v;
    __syncthreads();
    for (int off = 64; off > 0; off >>= 1) {
        if (c < off) { red[c] += red[c + off]; red2[c] += red2[c + off]; }
        __syncthreads();
    }
    if (c == 0) {
        float m = red[0] * (1.f / 128.f);
        float var = red2[0] * (1.f / 128.f) - m * m;
        mu_s = m; ris_s = rsqrtf(var + LN_EPS);
    }
    __syncthreads();
    float out = lrelu((v - mu_s) * ris_s * gamma[c] + beta[c], 0.01f);
    if (add_res) { out += x0[(size_t)n * 128 + c]; x0[(size_t)n * 128 + c] = out; }
    h[(size_t)n * 128 + c] = out;
    hh[(size_t)n * 128 + c] = (_Float16)out;
}

// ---------------------------------------------------------------------------
// Attention pooling + heads
// ---------------------------------------------------------------------------
__global__ void pool_init_k(unsigned* smax, float* wsum, float* gf)
{
    int t = threadIdx.x;
    if (t == 0) { *smax = fenc(-3.0e38f); *wsum = 0.f; }
    if (t < 128) gf[t] = 0.f;
}

__global__ __launch_bounds__(64)
void pool_score_k(const float* __restrict__ h, const float* __restrict__ w1,
                  const float* __restrict__ b1, const float* __restrict__ w2,
                  const float* __restrict__ b2, float* __restrict__ s,
                  unsigned* __restrict__ smax)
{
    __shared__ float red[64];
    int n = blockIdx.x, j = threadIdx.x;
    float acc = b1[j];
    for (int c = 0; c < 128; c++) acc += h[(size_t)n * 128 + c] * w1[c * 64 + j];
    red[j] = lrelu(acc, 0.01f) * w2[j];
    __syncthreads();
    for (int off = 32; off > 0; off >>= 1) {
        if (j < off) red[j] += red[j + off];
        __syncthreads();
    }
    if (j == 0) {
        float sv = red[0] + b2[0];
        s[n] = sv;
        atomicMax(smax, fenc(sv));
    }
}

__global__ void pool_exp_k(const float* __restrict__ s, const unsigned* __restrict__ smax,
                           float* __restrict__ ew, float* __restrict__ wsum)
{
    int n = blockIdx.x * blockDim.x + threadIdx.x;
    if (n >= N_NODES) return;
    float x = __expf(s[n] - fdec(*smax));
    ew[n] = x;
    atomicAdd(wsum, x);
}

__global__ __launch_bounds__(128)
void pool_accum_k(const float* __restrict__ h, const float* __restrict__ ew,
                  float* __restrict__ gf)
{
    int c = threadIdx.x;
    int n0 = blockIdx.x * 256;
    int n1 = n0 + 256; if (n1 > N_NODES) n1 = N_NODES;
    float acc = 0.f;
    for (int n = n0; n < n1; n++) acc += h[(size_t)n * 128 + c] * ew[n];
    atomicAdd(&gf[c], acc);
}

__device__ void block_ln_128(float* buf, const float* g, const float* b,
                             float slope, float* red, float* red2, float* stat)
{
    int t = threadIdx.x;
    if (t < 128) { red[t] = buf[t]; red2[t] = buf[t] * buf[t]; }
    __syncthreads();
    for (int off = 64; off > 0; off >>= 1) {
        if (t < off) { red[t] += red[t + off]; red2[t] += red2[t + off]; }
        __syncthreads();
    }
    if (t == 0) {
        float m = red[0] * (1.f / 128.f);
        float var = red2[0] * (1.f / 128.f) - m * m;
        stat[0] = m; stat[1] = rsqrtf(var + LN_EPS);
    }
    __syncthreads();
    if (t < 128) buf[t] = lrelu((buf[t] - stat[0]) * stat[1] * g[t] + b[t], slope);
    __syncthreads();
}

__global__ __launch_bounds__(256)
void final_head_k(const float* __restrict__ gf, const float* __restrict__ wsum,
                  const float* __restrict__ tf, const float* __restrict__ task_w,
                  const float* __restrict__ task_b, const float* __restrict__ task_g,
                  const float* __restrict__ task_be,
                  const float* __restrict__ v_w1, const float* __restrict__ v_b1,
                  const float* __restrict__ v_g, const float* __restrict__ v_be,
                  const float* __restrict__ v_w2, const float* __restrict__ v_b2,
                  const float* __restrict__ a_w1, const float* __restrict__ a_b1,
                  const float* __restrict__ a_g, const float* __restrict__ a_be,
                  const float* __restrict__ a_w2, const float* __restrict__ a_b2,
                  float* __restrict__ out)
{
    __shared__ float cf[256];
    __shared__ float buf[128];
    __shared__ float red[128], red2[128];
    __shared__ float stat[2];
    __shared__ float adv[16];
    __shared__ float value_s;
    int t = threadIdx.x;

    if (t < 128) {
        float acc = task_b[t];
        #pragma unroll
        for (int k = 0; k < 4; k++) acc += tf[k] * task_w[k * 128 + t];
        buf[t] = acc;
    }
    __syncthreads();
    block_ln_128(buf, task_g, task_be, 0.01f, red, red2, stat);
    if (t < 128) { cf[t] = gf[t] / (*wsum); cf[128 + t] = buf[t]; }
    __syncthreads();

    if (t < 128) {
        float acc = v_b1[t];
        for (int k = 0; k < 256; k++) acc += cf[k] * v_w1[k * 128 + t];
        buf[t] = acc;
    }
    __syncthreads();
    block_ln_128(buf, v_g, v_be, 0.01f, red, red2, stat);
    if (t < 128) red[t] = buf[t] * v_w2[t];
    __syncthreads();
    for (int off = 64; off > 0; off >>= 1) {
        if (t < off) red[t] += red[t + off];
        __syncthreads();
    }
    if (t == 0) value_s = red[0] + v_b2[0];
    __syncthreads();

    if (t < 128) {
        float acc = a_b1[t];
        for (int k = 0; k < 256; k++) acc += cf[k] * a_w1[k * 128 + t];
        buf[t] = acc;
    }
    __syncthreads();
    block_ln_128(buf, a_g, a_be, 0.01f, red, red2, stat);
    if (t < OUT_DIM) {
        float acc = a_b2[t];
        for (int c = 0; c < 128; c++) acc += buf[c] * a_w2[c * OUT_DIM + t];
        adv[t] = acc;
    }
    __syncthreads();
    if (t == 0) {
        float m = 0.f;
        for (int k = 0; k < OUT_DIM; k++) m += adv[k];
        m *= (1.f / OUT_DIM);
        for (int k = 0; k < OUT_DIM; k++) out[k] = value_s + adv[k] - m;
    }
}

// ---------------------------------------------------------------------------
extern "C" void kernel_launch(void* const* d_in, const int* in_sizes, int n_in,
                              void* d_out, int out_size, void* d_ws, size_t ws_size,
                              hipStream_t stream)
{
    const float* x        = (const float*)d_in[0];
    const int*   eidx     = (const int*)  d_in[1];
    const float* eattr    = (const float*)d_in[2];
    const float* tf       = (const float*)d_in[3];
    const float* emb_w    = (const float*)d_in[4];
    const float* emb_b    = (const float*)d_in[5];
    const float* emb_g    = (const float*)d_in[6];
    const float* emb_be   = (const float*)d_in[7];
    const float* gat_w    = (const float*)d_in[8];
    const float* gat_as   = (const float*)d_in[9];
    const float* gat_ad   = (const float*)d_in[10];
    const float* gat_bias = (const float*)d_in[11];
    const float* ec_w1    = (const float*)d_in[12];
    const float* ec_b1    = (const float*)d_in[13];
    const float* ec_g1    = (const float*)d_in[14];
    const float* ec_be1   = (const float*)d_in[15];
    const float* ec_w2    = (const float*)d_in[16];
    const float* ec_b2    = (const float*)d_in[17];
    const float* ec_g2    = (const float*)d_in[18];
    const float* ec_be2   = (const float*)d_in[19];
    const float* norm_g   = (const float*)d_in[20];
    const float* norm_b   = (const float*)d_in[21];
    const float* task_w   = (const float*)d_in[22];
    const float* task_b   = (const float*)d_in[23];
    const float* task_g   = (const float*)d_in[24];
    const float* task_be  = (const float*)d_in[25];
    const float* pa_w1    = (const float*)d_in[26];
    const float* pa_b1    = (const float*)d_in[27];
    const float* pa_w2    = (const float*)d_in[28];
    const float* pa_b2    = (const float*)d_in[29];
    const float* v_w1     = (const float*)d_in[30];
    const float* v_b1     = (const float*)d_in[31];
    const float* v_g      = (const float*)d_in[32];
    const float* v_be     = (const float*)d_in[33];
    const float* v_w2     = (const float*)d_in[34];
    const float* v_b2     = (const float*)d_in[35];
    const float* a_w1     = (const float*)d_in[36];
    const float* a_b1     = (const float*)d_in[37];
    const float* a_g      = (const float*)d_in[38];
    const float* a_be     = (const float*)d_in[39];
    const float* a_w2     = (const float*)d_in[40];
    const float* a_b2     = (const float*)d_in[41];

    const int* src = eidx;
    const int* dst = eidx + N_EDGES;

    const size_t NH = (size_t)N_NODES * HIDDIM;
    float* p = (float*)d_ws;
    float* h      = p; p += NH;
    float* x0     = p; p += NH;
    float* ht     = p; p += NH;
    float* x1     = p; p += NH;
    float* x2     = p; p += NH;
    float* a_s    = p; p += N_NODES * 2;
    float* a_d    = p; p += N_NODES * 2;
    float* e_self = p; p += N_NODES * 2;
    unsigned* emax = (unsigned*)p; p += N_NODES * 2;
    float* denom  = p; p += N_NODES * 2;
    float* e_val  = p; p += (size_t)N_EDGES * 2;
    float* ee     = p; p += (size_t)N_EDGES * 2;
    float* s_sc   = p; p += N_NODES;
    float* ew     = p; p += N_NODES;
    unsigned* smax = (unsigned*)p; p += 1;
    float* wsum   = p; p += 1;
    float* gf     = p; p += 128;
    // f16 region
    _Float16* hp = (_Float16*)p;
    _Float16* h_half = hp; hp += NH;
    _Float16* emb_wt = hp; hp += 128 * 128;
    _Float16* gat_wt = hp; hp += (size_t)LAYERS * 128 * 128;
    _Float16* ec1_wt = hp; hp += (size_t)LAYERS * 256 * 128;
    _Float16* ec2_wt = hp; hp += (size_t)LAYERS * 128 * 128;

    float* outp = (float*)d_out;

    const int nodeTiles = (N_NODES + 15) / 16;
    const int edgeTiles = N_EDGES / EPB;
    const int eh_blocks = (N_EDGES * 2 + 255) / 256;
    const int nh_blocks = (N_NODES * 2 + 255) / 256;

    // weight pre-convert (+transpose) to f16
    cvt_wt_k<<<(128 * 128 + 255) / 256, 256, 0, stream>>>(emb_w, emb_wt, 128, 128, 1);
    cvt_wt_k<<<(3 * 128 * 128 + 255) / 256, 256, 0, stream>>>(gat_w, gat_wt, 128, 128, 3);
    cvt_wt_k<<<(3 * 256 * 128 + 255) / 256, 256, 0, stream>>>(ec_w1, ec1_wt, 256, 128, 3);
    cvt_wt_k<<<(3 * 128 * 128 + 255) / 256, 256, 0, stream>>>(ec_w2, ec2_wt, 128, 128, 3);

    // Embedding: h = lrelu(LN(x @ emb_w + emb_b)), also f16 shadow
    node_gemm_k128<<<nodeTiles, 256, 0, stream>>>(x, nullptr, emb_wt, emb_b, emb_g, emb_be,
                                                  h, h_half, N_NODES, 1, 1, 0.01f);
    (void)hipMemcpyAsync(x0, h, NH * sizeof(float), hipMemcpyDeviceToDevice, stream);

    for (int i = 0; i < LAYERS; i++) {
        node_gemm_k128<<<nodeTiles, 256, 0, stream>>>(nullptr, h_half,
                                                      gat_wt + (size_t)i * 128 * 128,
                                                      nullptr, nullptr, nullptr,
                                                      ht, nullptr, N_NODES, 0, 0, 0.f);
        gat_prep_k<<<N_NODES, 128, 0, stream>>>(ht, gat_as + i * 128, gat_ad + i * 128,
                                                gat_bias + i * 128, a_s, a_d, e_self,
                                                emax, denom, x1, x2);
        gat_edge_max_k<<<eh_blocks, 256, 0, stream>>>(src, dst, a_s, a_d, e_val, emax);
        gat_edge_exp_k<<<eh_blocks, 256, 0, stream>>>(dst, e_val, emax, ee, denom);
        gat_self_k<<<nh_blocks, 256, 0, stream>>>(e_self, emax, denom);
        gat_scatter_k<<<N_EDGES, 128, 0, stream>>>(src, dst, ht, ee, denom, x1);

        edgeconv_k<<<edgeTiles, 256, 0, stream>>>(h_half, src, dst, eattr,
                                                  ec1_wt + (size_t)i * 256 * 128,
                                                  ec_b1 + i * 128, ec_g1 + i * 128, ec_be1 + i * 128,
                                                  ec2_wt + (size_t)i * 128 * 128,
                                                  ec_b2 + i * 128, ec_g2 + i * 128, ec_be2 + i * 128,
                                                  x2);

        combine_k<<<N_NODES, 128, 0, stream>>>(x1, x2, ht, e_self, denom,
                                               norm_g + i * 128, norm_b + i * 128,
                                               h, h_half, x0, i > 0 ? 1 : 0);
    }

    pool_init_k<<<1, 256, 0, stream>>>(smax, wsum, gf);
    pool_score_k<<<N_NODES, 64, 0, stream>>>(h, pa_w1, pa_b1, pa_w2, pa_b2, s_sc, smax);
    pool_exp_k<<<(N_NODES + 255) / 256, 256, 0, stream>>>(s_sc, smax, ew, wsum);
    pool_accum_k<<<(N_NODES + 255) / 256, 128, 0, stream>>>(h, ew, gf);

    final_head_k<<<1, 256, 0, stream>>>(gf, wsum, tf, task_w, task_b, task_g, task_be,
                                        v_w1, v_b1, v_g, v_be, v_w2, v_b2,
                                        a_w1, a_b1, a_g, a_be, a_w2, a_b2, outp);
}
